// WindowAttention3D_D_9964324126895
// MI455X (gfx1250) — matrix operand order repaired and checked by reading, not compiled, not run on hardware
//
#include <hip/hip_runtime.h>

// ---------- types / helpers -------------------------------------------------

typedef __attribute__((ext_vector_type(16))) __bf16 v16bf;
typedef __attribute__((ext_vector_type(8)))  float  v8f;

union FragU {
  int4  q[2];
  v16bf v;
};

__device__ __forceinline__ unsigned short f2bf(float f) {
  unsigned int u = __float_as_uint(f);
  u += 0x7FFFu + ((u >> 16) & 1u);   // round-to-nearest-even
  return (unsigned short)(u >> 16);
}

// A-matrix fragment (16x32 bf16) from a row-major [16][ldk] bf16 tile.
// Lane L: row = L&15; elems 0..7  <-> K = (L>>4)*8 .. +7
//                     elems 8..15 <-> K = 16+(L>>4)*8 .. +7
__device__ __forceinline__ v16bf load_a_frag(const unsigned short* tile, int ldk, int lane) {
  const int m  = lane & 15;
  const int kh = (lane >> 4) << 3;
  const unsigned short* p = tile + m * ldk + kh;
  FragU f;
  f.q[0] = *(const int4*)(p);
  f.q[1] = *(const int4*)(p + 16);
  return f.v;
}

// B-matrix fragment (32x16 bf16) from a row-major [32][ldn] tile.
// Lane L: col = L&15; elems 0..7 <-> K = (L>>4)*8 .. +7, elems 8..15 <-> K = 16+(L>>4)*8 .. +7
__device__ __forceinline__ v16bf load_b_frag(const unsigned short* p, int ldn, int lane) {
  union { unsigned short u[16]; v16bf v; } f;
  const int n = lane & 15, kh = (lane >> 4) << 3;
#pragma unroll
  for (int s = 0; s < 16; ++s) {
    int k = (s < 8) ? (kh + s) : (16 + kh + s - 8);
    f.u[s] = p[k * ldn + n];
  }
  return f.v;
}

#define WMMA_BF16(A, B, C) \
  __builtin_amdgcn_wmma_f32_16x16x32_bf16(false, (A), false, (B), (short)0, (C), false, false)

// ---------- CDNA5 async global->LDS staging (ASYNCcnt-tracked DMA) ----------

#define USE_ASYNC_LDS 1

__device__ __forceinline__ void async_copy16(void* lds_dst, const void* gsrc) {
#if USE_ASYNC_LDS
  unsigned loff = (unsigned)(uintptr_t)lds_dst;              // LDS byte offset
  unsigned long long ga = (unsigned long long)(uintptr_t)gsrc;
  asm volatile("global_load_async_to_lds_b128 %0, %1, off"
               :: "v"(loff), "v"(ga) : "memory");
#else
  *(int4*)lds_dst = *(const int4*)gsrc;
#endif
}

__device__ __forceinline__ void async_wait() {
#if USE_ASYNC_LDS
  asm volatile("s_wait_asynccnt 0x0" ::: "memory");
#endif
}

// ---------- constants -------------------------------------------------------
// B_=512, N=256, C=192, H=6, hd=32, nW=64, table=1575
#define QK_SCALE 0.17677669529663687f   // 32^-0.5

// ---------- kernel 1: x (fp32) -> xbf (bf16) --------------------------------

__global__ __launch_bounds__(256) void k_cvt_x(const float* __restrict__ x,
                                               unsigned short* __restrict__ xbf, int n8) {
  int i = blockIdx.x * 256 + threadIdx.x;
  if (i >= n8) return;
  const float4* px = (const float4*)x + (size_t)i * 2;
  float4 a = px[0], b = px[1];
  int4 o;
  o.x = (int)((unsigned)f2bf(a.x) | ((unsigned)f2bf(a.y) << 16));
  o.y = (int)((unsigned)f2bf(a.z) | ((unsigned)f2bf(a.w) << 16));
  o.z = (int)((unsigned)f2bf(b.x) | ((unsigned)f2bf(b.y) << 16));
  o.w = (int)((unsigned)f2bf(b.z) | ((unsigned)f2bf(b.w) << 16));
  ((int4*)xbf)[i] = o;
}

// ---------- kernel 2: transpose+convert weights -----------------------------

__global__ __launch_bounds__(256) void k_prep_w(const float* __restrict__ qkv_w,
                                                const float* __restrict__ proj_w,
                                                unsigned short* __restrict__ WTq,
                                                unsigned short* __restrict__ WTp) {
  int i = blockIdx.x * 256 + threadIdx.x;
  if (i < 576 * 192) {
    int n = i / 192, k = i % 192;
    WTq[k * 576 + n] = f2bf(qkv_w[i]);
  }
  if (i < 192 * 192) {
    int n = i / 192, k = i % 192;
    WTp[k * 192 + n] = f2bf(proj_w[i]);
  }
}

// ---------- kernel 3: bias gather  bias[h][n][m] = rpb_table[rpi[n][m]][h] --

__global__ __launch_bounds__(256) void k_bias(const int* __restrict__ rpi,
                                              const float* __restrict__ tab,
                                              float* __restrict__ biasHNN) {
  int i = blockIdx.x * 256 + threadIdx.x;   // 0..65535 == n*256+m
  int idx = rpi[i];
#pragma unroll
  for (int h = 0; h < 6; ++h)
    biasHNN[h * 65536 + i] = tab[idx * 6 + h];
}

// ---------- kernel 4: QKV GEMM (131072x576x192, bf16 WMMA) ------------------
// One block = 64 M-rows. A-panel (64x192, contiguous 24KB) staged to LDS ONCE
// and cached as 6 A-fragments in VGPRs; loop 9 N-tiles of the L2-resident
// weight panel. Epilogue scatters Q (pre-scaled) / KT / V bf16 head layouts.

__global__ __launch_bounds__(256) void k_qkv_gemm(const unsigned short* __restrict__ xbf,
                                                  const unsigned short* __restrict__ WTq,
                                                  const float* __restrict__ qkv_b,
                                                  unsigned short* __restrict__ Qbf,
                                                  unsigned short* __restrict__ KTbf,
                                                  unsigned short* __restrict__ Vbf) {
  __shared__ unsigned short As[64 * 192];   // 24KB
  __shared__ unsigned short Bs[192 * 64];   // 24KB
  const int tid = threadIdx.x;
  const int mt = blockIdx.x;
  const int wid = tid >> 5, lane = tid & 31;
  const int msub = wid & 3, nsub = wid >> 2;
  const int hlf = lane >> 4, l15 = lane & 15;
  const v8f zero = {0, 0, 0, 0, 0, 0, 0, 0};

  // stage A (contiguous 1536 x 16B)
  {
    const unsigned short* ga = xbf + (size_t)mt * (64 * 192);
    for (int i = tid; i < 1536; i += 256)
      async_copy16(&As[i * 8], ga + i * 8);
  }

  v16bf afr[6];
  for (int nt = 0; nt < 9; ++nt) {
    // stage B tile: Bs[k][n'] = WTq[k][nt*64+n']
    for (int i = tid; i < 1536; i += 256) {
      int k = i >> 3, seg = (i & 7) << 3;
      async_copy16(&Bs[k * 64 + seg], WTq + (size_t)k * 576 + nt * 64 + seg);
    }
    async_wait();
    __syncthreads();
    if (nt == 0) {
#pragma unroll
      for (int k0 = 0; k0 < 6; ++k0)
        afr[k0] = load_a_frag(As + msub * (16 * 192) + k0 * 32, 192, lane);
    }
    v8f acc0 = zero, acc1 = zero;
#pragma unroll
    for (int k0 = 0; k0 < 6; ++k0) {
      v16bf b0 = load_b_frag(Bs + (k0 * 32) * 64 + nsub * 32, 64, lane);
      v16bf b1 = load_b_frag(Bs + (k0 * 32) * 64 + nsub * 32 + 16, 64, lane);
      acc0 = WMMA_BF16(afr[k0], b0, acc0);
      acc1 = WMMA_BF16(afr[k0], b1, acc1);
    }
    // epilogue scatter
#pragma unroll
    for (int jj = 0; jj < 2; ++jj) {
      v8f acc = jj ? acc1 : acc0;
      int n = nt * 64 + nsub * 32 + jj * 16 + l15;
      float bv = qkv_b[n];
      int sec = n / 192, c = n % 192;
      int h = c >> 5, d = c & 31;
#pragma unroll
      for (int r = 0; r < 8; ++r) {
        int m = mt * 64 + msub * 16 + r + 8 * hlf;
        int b = m >> 8, nn = m & 255;
        float v = acc[r] + bv;
        size_t hid = (size_t)(b * 6 + h) * 8192;
        if (sec == 0)      Qbf[hid + nn * 32 + d]           = f2bf(v * QK_SCALE);
        else if (sec == 1) KTbf[hid + (size_t)d * 256 + nn] = f2bf(v);
        else               Vbf[hid + nn * 32 + d]           = f2bf(v);
      }
    }
    __syncthreads();   // protect Bs before restage
  }
}

// ---------- kernel 5: fused attention per (b,h) -----------------------------

__global__ __launch_bounds__(256) void k_attn(const unsigned short* __restrict__ Qbf,
                                              const unsigned short* __restrict__ KTbf,
                                              const unsigned short* __restrict__ Vbf,
                                              const float* __restrict__ biasHNN,
                                              const float* __restrict__ mask,
                                              unsigned short* __restrict__ Obf) {
  __shared__ unsigned short Qs[256 * 32];
  __shared__ unsigned short KTs[32 * 256];
  __shared__ unsigned short Vs[256 * 32];
  __shared__ unsigned short Ps[8 * 16 * 32];
  const int tid  = threadIdx.x;
  const int head = blockIdx.x;            // b*6 + h
  const int b = head / 6, h = head % 6;
  const int w = b & 63;
  const size_t hb = (size_t)head * 8192;
  for (int i = tid; i < 1024; i += 256) {
    async_copy16(&Qs[i * 8],  Qbf  + hb + i * 8);
    async_copy16(&KTs[i * 8], KTbf + hb + i * 8);
    async_copy16(&Vs[i * 8],  Vbf  + hb + i * 8);
  }
  async_wait();
  __syncthreads();
  const int wid = tid >> 5, lane = tid & 31;
  const int hlf = lane >> 4, l15 = lane & 15;
  const float* biasH = biasHNN + (size_t)h * 65536;
  const float* maskW = mask + (size_t)w * 65536;
  unsigned short* Pw = Ps + wid * (16 * 32);
  const v8f zero = {0, 0, 0, 0, 0, 0, 0, 0};

  for (int mtile = wid; mtile < 16; mtile += 8) {
    v8f S[16];
#pragma unroll
    for (int j = 0; j < 16; ++j) S[j] = zero;
    // S = Q K^T  (K-dim = hd = 32 -> one WMMA per 16x16 tile)
    v16bf aq = load_a_frag(Qs + mtile * (16 * 32), 32, lane);
#pragma unroll
    for (int j = 0; j < 16; ++j) {
      v16bf bk = load_b_frag(KTs + j * 16, 256, lane);
      S[j] = WMMA_BF16(aq, bk, S[j]);
    }
    // + relative-position bias + window mask
#pragma unroll
    for (int j = 0; j < 16; ++j) {
      int col = j * 16 + l15;
#pragma unroll
      for (int r = 0; r < 8; ++r) {
        int q = mtile * 16 + r + 8 * hlf;
        S[j][r] += biasH[q * 256 + col] + maskW[q * 256 + col];
      }
    }
    // softmax over 256 cols: each row lives in one 16-lane half x 16 frags
#pragma unroll
    for (int r = 0; r < 8; ++r) {
      float mx = -3.4e38f;
#pragma unroll
      for (int j = 0; j < 16; ++j) mx = fmaxf(mx, S[j][r]);
      mx = fmaxf(mx, __shfl_xor(mx, 1, 32));
      mx = fmaxf(mx, __shfl_xor(mx, 2, 32));
      mx = fmaxf(mx, __shfl_xor(mx, 4, 32));
      mx = fmaxf(mx, __shfl_xor(mx, 8, 32));
      float sum = 0.f;
#pragma unroll
      for (int j = 0; j < 16; ++j) { float e = __expf(S[j][r] - mx); S[j][r] = e; sum += e; }
      sum += __shfl_xor(sum, 1, 32);
      sum += __shfl_xor(sum, 2, 32);
      sum += __shfl_xor(sum, 4, 32);
      sum += __shfl_xor(sum, 8, 32);
      float inv = 1.0f / sum;
#pragma unroll
      for (int j = 0; j < 16; ++j) S[j][r] *= inv;
    }
    // O = P @ V : per-wave LDS transpose (C-layout -> A-layout), 2 WMMA / kk
    v8f O0 = zero, O1 = zero;
    for (int kk = 0; kk < 8; ++kk) {
#pragma unroll
      for (int jj = 0; jj < 2; ++jj) {
        int j = kk * 2 + jj;
#pragma unroll
        for (int r = 0; r < 8; ++r)
          Pw[(r + 8 * hlf) * 32 + jj * 16 + l15] = f2bf(S[j][r]);
      }
      v16bf ap  = load_a_frag(Pw, 32, lane);
      v16bf bv0 = load_b_frag(Vs + (kk * 32) * 32, 32, lane);
      v16bf bv1 = load_b_frag(Vs + (kk * 32) * 32 + 16, 32, lane);
      O0 = WMMA_BF16(ap, bv0, O0);
      O1 = WMMA_BF16(ap, bv1, O1);
    }
    // store O as bf16 into [b][n][h*32+d] for the proj GEMM
#pragma unroll
    for (int r = 0; r < 8; ++r) {
      int q = mtile * 16 + r + 8 * hlf;
      size_t o = ((size_t)(b * 256 + q)) * 192 + h * 32;
      Obf[o + l15]      = f2bf(O0[r]);
      Obf[o + 16 + l15] = f2bf(O1[r]);
    }
  }
}

// ---------- kernel 6: proj GEMM (131072x192x192) + bias -> fp32 out ---------
// Same persistent-A structure: stage 64x192 A once, cache 6 A-frags, 3 N-tiles.

__global__ __launch_bounds__(256) void k_proj_gemm(const unsigned short* __restrict__ Obf,
                                                   const unsigned short* __restrict__ WTp,
                                                   const float* __restrict__ proj_b,
                                                   float* __restrict__ out) {
  __shared__ unsigned short As[64 * 192];
  __shared__ unsigned short Bs[192 * 64];
  const int tid = threadIdx.x;
  const int mt = blockIdx.x;
  const int wid = tid >> 5, lane = tid & 31;
  const int msub = wid & 3, nsub = wid >> 2;
  const int hlf = lane >> 4, l15 = lane & 15;
  const v8f zero = {0, 0, 0, 0, 0, 0, 0, 0};

  {
    const unsigned short* ga = Obf + (size_t)mt * (64 * 192);
    for (int i = tid; i < 1536; i += 256)
      async_copy16(&As[i * 8], ga + i * 8);
  }

  v16bf afr[6];
  for (int nt = 0; nt < 3; ++nt) {
    for (int i = tid; i < 1536; i += 256) {
      int k = i >> 3, seg = (i & 7) << 3;
      async_copy16(&Bs[k * 64 + seg], WTp + (size_t)k * 192 + nt * 64 + seg);
    }
    async_wait();
    __syncthreads();
    if (nt == 0) {
#pragma unroll
      for (int k0 = 0; k0 < 6; ++k0)
        afr[k0] = load_a_frag(As + msub * (16 * 192) + k0 * 32, 192, lane);
    }
    v8f acc0 = zero, acc1 = zero;
#pragma unroll
    for (int k0 = 0; k0 < 6; ++k0) {
      v16bf b0 = load_b_frag(Bs + (k0 * 32) * 64 + nsub * 32, 64, lane);
      v16bf b1 = load_b_frag(Bs + (k0 * 32) * 64 + nsub * 32 + 16, 64, lane);
      acc0 = WMMA_BF16(afr[k0], b0, acc0);
      acc1 = WMMA_BF16(afr[k0], b1, acc1);
    }
#pragma unroll
    for (int jj = 0; jj < 2; ++jj) {
      v8f acc = jj ? acc1 : acc0;
      int n = nt * 64 + nsub * 32 + jj * 16 + l15;
      float bv = proj_b[n];
#pragma unroll
      for (int r = 0; r < 8; ++r) {
        int m = mt * 64 + msub * 16 + r + 8 * hlf;
        out[(size_t)m * 192 + n] = acc[r] + bv;
      }
    }
    __syncthreads();
  }
}

// ---------- launcher --------------------------------------------------------

extern "C" void kernel_launch(void* const* d_in, const int* in_sizes, int n_in,
                              void* d_out, int out_size, void* d_ws, size_t ws_size,
                              hipStream_t stream) {
  (void)in_sizes; (void)n_in; (void)out_size; (void)ws_size;
  const float* x      = (const float*)d_in[0];
  const int*   rpi    = (const int*)d_in[1];
  const float* mask   = (const float*)d_in[2];
  const float* tab    = (const float*)d_in[3];
  const float* qkv_w  = (const float*)d_in[4];
  const float* qkv_b  = (const float*)d_in[5];
  const float* proj_w = (const float*)d_in[6];
  const float* proj_b = (const float*)d_in[7];
  float* out = (float*)d_out;
  char* ws = (char*)d_ws;

  // workspace carve-up (bytes, all 256-aligned)
  unsigned short* xbf  = (unsigned short*)(ws + 0);           // 48MB
  unsigned short* Obf  = (unsigned short*)(ws + 50331648);    // 48MB
  unsigned short* Qbf  = (unsigned short*)(ws + 100663296);   // 48MB
  unsigned short* KTbf = (unsigned short*)(ws + 150994944);   // 48MB
  unsigned short* Vbf  = (unsigned short*)(ws + 201326592);   // 48MB
  unsigned short* WTq  = (unsigned short*)(ws + 251658240);   // 192*576*2
  unsigned short* WTp  = (unsigned short*)(ws + 251658240 + 221184); // 192*192*2
  float*          biasHNN = (float*)(ws + 251658240 + 221184 + 73728); // 6*65536*4

  k_cvt_x<<<12288, 256, 0, stream>>>(x, xbf, 3145728);
  k_prep_w<<<432, 256, 0, stream>>>(qkv_w, proj_w, WTq, WTp);
  k_bias<<<256, 256, 0, stream>>>(rpi, tab, biasHNN);
  k_qkv_gemm<<<2048, 256, 0, stream>>>(xbf, WTq, qkv_b, Qbf, KTbf, Vbf);
  k_attn<<<3072, 256, 0, stream>>>(Qbf, KTbf, Vbf, biasHNN, mask, Obf);
  k_proj_gemm<<<2048, 256, 0, stream>>>(Obf, WTp, proj_b, out);
}
